// OperationMachineAttention_57939108823288
// MI455X (gfx1250) — matrix operand order repaired and checked by reading, not compiled, not run on hardware
//
#include <hip/hip_runtime.h>
#include <hip/hip_bf16.h>

typedef __attribute__((ext_vector_type(16))) _Float16 v16h;
typedef __attribute__((ext_vector_type(8)))  _Float16 v8h;
typedef __attribute__((ext_vector_type(4)))  _Float16 v4h;
typedef __attribute__((ext_vector_type(8)))  float    v8f;

#define Bq   8
#define Mq   1024
#define Cq   4096
#define Hq   128     // HID == OP_DIM == MACH_DIM == 128
#define LEAK 0.01f

// ---- workspace layout (bytes) ----
#define OFF_VT    0u               // f16 vT[b][h][c]      : 8*128*4096*2 = 8388608
#define OFF_SQ    8388608u         // f32 sq[b*M]          : 32768
#define OFF_SK    8421376u         // f32 sk[b*C]          : 131072
#define OFF_WQ    8552448u         // f32 wq_eff[128]      : 512
#define OFF_WK    8552960u         // f32 wk_eff[128]      : 512
#define OFF_BIAS  8553472u         // f32 biases[2]        : pad to 256
#define OFF_WTV   8553728u         // f16 wTv[h][k]        : 32768
#define OFF_OWT   8586496u         // f16 owT[n][k]        : 32768

static __device__ inline v16h combine16(v8h a0, v8h a1) {
  return __builtin_shufflevector(a0, a1, 0,1,2,3,4,5,6,7,8,9,10,11,12,13,14,15);
}

// ---------------------------------------------------------------------------
// prep: fold score_w into effective vectors; build f16 transposed weights.
// wq_eff[k] = sum_h q_w[k,h]*score_w[h];  bq = q_b.wq + score_b
// wk_eff[d] = sum_h kv_w[d,h]*score_w[128+h];  bk = kv_b[:128].wk
// wTv[h][k] = (f16) kv_w[k][128+h];  owT[n][k] = (f16) out_w[k][n]
// ---------------------------------------------------------------------------
__global__ void prep_kernel(const float* __restrict__ qw, const float* __restrict__ qb,
                            const float* __restrict__ kvw, const float* __restrict__ kvb,
                            const float* __restrict__ scw, const float* __restrict__ scb,
                            const float* __restrict__ ow,
                            float* __restrict__ wq_eff, float* __restrict__ wk_eff,
                            float* __restrict__ biases,
                            _Float16* __restrict__ wTv, _Float16* __restrict__ owT) {
  const int t = threadIdx.x;  // 256 threads
  if (t < 128) {
    float acc = 0.f;
    for (int h = 0; h < 128; ++h) acc += qw[t*128 + h] * scw[h];
    wq_eff[t] = acc;
  } else {
    const int d = t - 128;
    float acc = 0.f;
    for (int h = 0; h < 128; ++h) acc += kvw[d*256 + h] * scw[128 + h];
    wk_eff[d] = acc;
  }
  if (t == 0) {
    float bq = scb[0];
    for (int h = 0; h < 128; ++h) bq += qb[h] * scw[h];
    biases[0] = bq;
    float bk = 0.f;
    for (int h = 0; h < 128; ++h) bk += kvb[h] * scw[128 + h];
    biases[1] = bk;
  }
  for (int idx = t; idx < 128*128; idx += 256) {
    const int n = idx >> 7, k = idx & 127;
    wTv[idx] = (_Float16)kvw[k*256 + 128 + n];
    owT[idx] = (_Float16)ow[k*128 + n];
  }
}

// ---------------------------------------------------------------------------
// rowdot: outv[r] = rows[r,:128] . wv + biases[bias_idx]
// ---------------------------------------------------------------------------
__global__ void rowdot_kernel(const float* __restrict__ rows, const float* __restrict__ wv,
                              const float* __restrict__ biases, int bias_idx,
                              float* __restrict__ outv, int nrows) {
  const int r = blockIdx.x * blockDim.x + threadIdx.x;
  if (r >= nrows) return;
  const float4* rp = (const float4*)(rows + (size_t)r * 128);
  const float4* wp = (const float4*)wv;
  float acc = biases[bias_idx];
#pragma unroll 8
  for (int i = 0; i < 32; ++i) {
    const float4 a = rp[i], w = wp[i];
    acc += a.x*w.x + a.y*w.y + a.z*w.z + a.w*w.w;
  }
  outv[r] = acc;
}

// ---------------------------------------------------------------------------
// vproj: vT[b][n][c] = (f16)( opfeat[b,c,:] . kv_w[:,128+n] + kv_b[128+n] )
// One block = 16 rows (c) x 128 cols (n); 4 waves, each owns 32 cols.
// Output stored TRANSPOSED so attn's B fragments are contiguous 32-byte loads.
// ---------------------------------------------------------------------------
__global__ void vproj_kernel(const float* __restrict__ opfeat, const _Float16* __restrict__ wTv,
                             const float* __restrict__ kvb, _Float16* __restrict__ vT) {
  const int tid = threadIdx.x;
  const int w = tid >> 5, l = tid & 31, g = l >> 4, lr = l & 15;
  const int r16   = blockIdx.x * 16;     // flat row base in [0, B*C)
  const int b     = r16 >> 12;           // / C
  const int cbase = r16 & (Cq - 1);
  v8f acc0 = {}, acc1 = {};
  const float* arow = opfeat + (size_t)(r16 + lr) * 128;
  const int n0 = w*32 + lr, n1 = n0 + 16;
#pragma unroll
  for (int kk = 0; kk < 128; kk += 32) {
    // A fragment: row M=lr, K = g*16 .. g*16+15
    v16h a;
    {
      const float4* p0 = (const float4*)(arow + kk + g*16);
      const float4* p1 = (const float4*)(arow + kk + g*16 + 8);
      const float4 x0 = p0[0], x1 = p0[1], y0 = p1[0], y1 = p1[1];
      a[0]=(_Float16)x0.x; a[1]=(_Float16)x0.y; a[2]=(_Float16)x0.z; a[3]=(_Float16)x0.w;
      a[4]=(_Float16)x1.x; a[5]=(_Float16)x1.y; a[6]=(_Float16)x1.z; a[7]=(_Float16)x1.w;
      a[8]=(_Float16)y0.x; a[9]=(_Float16)y0.y; a[10]=(_Float16)y0.z; a[11]=(_Float16)y0.w;
      a[12]=(_Float16)y1.x; a[13]=(_Float16)y1.y; a[14]=(_Float16)y1.z; a[15]=(_Float16)y1.w;
    }
    // B fragment: col N=lr, K = g*16 .. g*16+15  (contiguous in wTv[n][k])
    const v16h b0 = *(const v16h*)(wTv + n0*128 + kk + g*16);
    const v16h b1 = *(const v16h*)(wTv + n1*128 + kk + g*16);
    acc0 = __builtin_amdgcn_wmma_f32_16x16x32_f16(false, a, false, b0, (short)0, acc0, false, false);
    acc1 = __builtin_amdgcn_wmma_f32_16x16x32_f16(false, a, false, b1, (short)0, acc1, false, false);
  }
  // D element i of lane(g,lr): (c = cbase + i + 8g, n).  8 consecutive c -> b128 store.
  {
    const float bias0 = kvb[128 + n0];
    v8h h0;
#pragma unroll
    for (int i = 0; i < 8; ++i) h0[i] = (_Float16)(acc0[i] + bias0);
    *(v8h*)(vT + ((size_t)b*Hq + n0)*Cq + cbase + g*8) = h0;
    const float bias1 = kvb[128 + n1];
    v8h h1;
#pragma unroll
    for (int i = 0; i < 8; ++i) h1[i] = (_Float16)(acc1[i] + bias1);
    *(v8h*)(vT + ((size_t)b*Hq + n1)*Cq + cbase + g*8) = h1;
  }
}

// ---------------------------------------------------------------------------
// attn: fused  leaky(sq+sk)*mask -> online softmax -> @ v -> @ out_w + out_b
// Block = (b, 16-row m-tile).  4 waves x 32 h-columns each.  C chunk = 64:
// each chunk does 4 WMMAs/wave with only 2 barriers.
// ---------------------------------------------------------------------------
__global__ void __launch_bounds__(128)
attn_kernel(const float* __restrict__ mask, const float* __restrict__ sq,
            const float* __restrict__ sk, const _Float16* __restrict__ vT,
            const _Float16* __restrict__ owT, const float* __restrict__ outb,
            float* __restrict__ out) {
  __shared__ float sqs[16], rowmax[16], rowscale[16], rowsum[16];
  __shared__ _Float16 pTile[16*64];    // A operand staging (f16, [m][c]) : 2KB
  __shared__ _Float16 wTile[16*128];   // weighted staging  (f16, [m][h]) : 4KB

  const int tid = threadIdx.x;
  const int w = tid >> 5, l = tid & 31, g = l >> 4, lr = l & 15;
  const int b  = blockIdx.x >> 6;         // 64 m-tiles per batch
  const int m0 = (blockIdx.x & 63) * 16;

  if (tid < 16) {
    sqs[tid]    = sq[b*Mq + m0 + tid];
    rowmax[tid] = -3.0e38f;
    rowsum[tid] = 0.0f;
  }
  __syncthreads();

  const int  mrow   = w*4 + (l >> 3);     // stats row owned by this thread
  const int  cl     = (l & 7) * 4;        // 4 consecutive c per thread (x2 halves)
  const bool leader = (l & 7) == 0;
  const float* mrowp = mask + ((size_t)b*Mq + m0 + mrow) * Cq;
  const float* skb   = sk + b*Cq;
  const _Float16* vb = vT + (size_t)b*Hq*Cq;
  const int n0 = w*32 + lr, n1 = n0 + 16;

  v8f acc0 = {}, acc1 = {};

  for (int c0 = 0; c0 < Cq; c0 += 64) {
    // prefetch mask stream ahead (global_prefetch_b8)
    if (c0 + 512 < Cq) __builtin_prefetch(mrowp + c0 + 512, 0, 1);

    // ---- scores for this 16x64 chunk (8 c-values per thread) ----
    const float4 mkA = *(const float4*)(mrowp + c0 + cl);
    const float4 mkB = *(const float4*)(mrowp + c0 + 32 + cl);
    const float4 skA = *(const float4*)(skb + c0 + cl);
    const float4 skB = *(const float4*)(skb + c0 + 32 + cl);
    const float sqm = sqs[mrow];
    float t0 = sqm + skA.x; t0 = (t0 >= 0.f ? t0 : LEAK*t0) * mkA.x;
    float t1 = sqm + skA.y; t1 = (t1 >= 0.f ? t1 : LEAK*t1) * mkA.y;
    float t2 = sqm + skA.z; t2 = (t2 >= 0.f ? t2 : LEAK*t2) * mkA.z;
    float t3 = sqm + skA.w; t3 = (t3 >= 0.f ? t3 : LEAK*t3) * mkA.w;
    float t4 = sqm + skB.x; t4 = (t4 >= 0.f ? t4 : LEAK*t4) * mkB.x;
    float t5 = sqm + skB.y; t5 = (t5 >= 0.f ? t5 : LEAK*t5) * mkB.y;
    float t6 = sqm + skB.z; t6 = (t6 >= 0.f ? t6 : LEAK*t6) * mkB.z;
    float t7 = sqm + skB.w; t7 = (t7 >= 0.f ? t7 : LEAK*t7) * mkB.w;

    // ---- online softmax: row max over 8 lanes ----
    float lmax = fmaxf(fmaxf(fmaxf(t0, t1), fmaxf(t2, t3)),
                       fmaxf(fmaxf(t4, t5), fmaxf(t6, t7)));
#pragma unroll
    for (int off = 1; off < 8; off <<= 1)
      lmax = fmaxf(lmax, __shfl_xor(lmax, off, 32));
    const float old = rowmax[mrow];            // uniform within 8-lane group
    const float nm  = fmaxf(old, lmax);
    const float sc  = __expf(old - nm);
    if (leader) { rowscale[mrow] = sc; rowmax[mrow] = nm; }

    const float p0 = __expf(t0 - nm), p1 = __expf(t1 - nm);
    const float p2 = __expf(t2 - nm), p3 = __expf(t3 - nm);
    const float p4 = __expf(t4 - nm), p5 = __expf(t5 - nm);
    const float p6 = __expf(t6 - nm), p7 = __expf(t7 - nm);
    float lsum = (p0 + p1 + p2 + p3) + (p4 + p5 + p6 + p7);
#pragma unroll
    for (int off = 1; off < 8; off <<= 1)
      lsum += __shfl_xor(lsum, off, 32);
    if (leader) rowsum[mrow] = rowsum[mrow] * sc + lsum;

    // ---- stage P (f16) ----
    v4h phA; phA[0]=(_Float16)p0; phA[1]=(_Float16)p1; phA[2]=(_Float16)p2; phA[3]=(_Float16)p3;
    v4h phB; phB[0]=(_Float16)p4; phB[1]=(_Float16)p5; phB[2]=(_Float16)p6; phB[3]=(_Float16)p7;
    *(v4h*)(&pTile[mrow*64 + cl]) = phA;
    *(v4h*)(&pTile[mrow*64 + 32 + cl]) = phB;
    __syncthreads();

    // ---- rescale accumulators (acc element i <-> row i+8g) ----
#pragma unroll
    for (int i = 0; i < 8; ++i) {
      const float s = rowscale[i + 8*g];
      acc0[i] *= s; acc1[i] *= s;
    }
    // ---- A from LDS, B from global vT, 4 WMMAs ----
    const v16h aA = combine16(*(const v8h*)(&pTile[lr*64 + g*16]),
                              *(const v8h*)(&pTile[lr*64 + g*16 + 8]));
    const v16h aB = combine16(*(const v8h*)(&pTile[lr*64 + 32 + g*16]),
                              *(const v8h*)(&pTile[lr*64 + 40 + g*16]));
    const v16h b0A = *(const v16h*)(vb + (size_t)n0*Cq + c0 + g*16);
    const v16h b0B = *(const v16h*)(vb + (size_t)n0*Cq + c0 + 32 + g*16);
    const v16h b1A = *(const v16h*)(vb + (size_t)n1*Cq + c0 + g*16);
    const v16h b1B = *(const v16h*)(vb + (size_t)n1*Cq + c0 + 32 + g*16);
    acc0 = __builtin_amdgcn_wmma_f32_16x16x32_f16(false, aA, false, b0A, (short)0, acc0, false, false);
    acc0 = __builtin_amdgcn_wmma_f32_16x16x32_f16(false, aB, false, b0B, (short)0, acc0, false, false);
    acc1 = __builtin_amdgcn_wmma_f32_16x16x32_f16(false, aA, false, b1A, (short)0, acc1, false, false);
    acc1 = __builtin_amdgcn_wmma_f32_16x16x32_f16(false, aB, false, b1B, (short)0, acc1, false, false);
    __syncthreads();
  }

  // ---- divide by row sums ----
  if (tid < 16) rowscale[tid] = 1.0f / rowsum[tid];
  __syncthreads();
#pragma unroll
  for (int i = 0; i < 8; ++i) {
    const float s = rowscale[i + 8*g];
    acc0[i] *= s; acc1[i] *= s;
  }
  // ---- stage weighted (f16) ----
#pragma unroll
  for (int i = 0; i < 8; ++i) {
    wTile[(i + 8*g)*128 + n0] = (_Float16)acc0[i];
    wTile[(i + 8*g)*128 + n1] = (_Float16)acc1[i];
  }
  __syncthreads();

  // ---- final projection: out = weighted @ out_w + out_b ----
  v8f o0 = {}, o1 = {};
#pragma unroll
  for (int kk = 0; kk < 128; kk += 32) {
    const v16h a = combine16(*(const v8h*)(&wTile[lr*128 + kk + g*16]),
                             *(const v8h*)(&wTile[lr*128 + kk + g*16 + 8]));
    const v16h bb0 = *(const v16h*)(owT + n0*128 + kk + g*16);
    const v16h bb1 = *(const v16h*)(owT + n1*128 + kk + g*16);
    o0 = __builtin_amdgcn_wmma_f32_16x16x32_f16(false, a, false, bb0, (short)0, o0, false, false);
    o1 = __builtin_amdgcn_wmma_f32_16x16x32_f16(false, a, false, bb1, (short)0, o1, false, false);
  }
  const float ob0 = outb[n0], ob1 = outb[n1];
  float* orow = out + ((size_t)b*Mq + m0) * 128;
#pragma unroll
  for (int i = 0; i < 8; ++i) {
    orow[(size_t)(i + 8*g)*128 + n0] = o0[i] + ob0;
    orow[(size_t)(i + 8*g)*128 + n1] = o1[i] + ob1;
  }
}

// ---------------------------------------------------------------------------
extern "C" void kernel_launch(void* const* d_in, const int* in_sizes, int n_in,
                              void* d_out, int out_size, void* d_ws, size_t ws_size,
                              hipStream_t stream) {
  const float* opfeat   = (const float*)d_in[0];   // (B,C,128)
  const float* machfeat = (const float*)d_in[1];   // (B,M,128)
  const float* mask     = (const float*)d_in[2];   // (B,M,C)
  const float* qw       = (const float*)d_in[3];   // (128,128)
  const float* qb       = (const float*)d_in[4];   // (128,)
  const float* kvw      = (const float*)d_in[5];   // (128,256)
  const float* kvb      = (const float*)d_in[6];   // (256,)
  const float* scw      = (const float*)d_in[7];   // (256,1)
  const float* scb      = (const float*)d_in[8];   // (1,)
  const float* ow       = (const float*)d_in[9];   // (128,128)
  const float* ob       = (const float*)d_in[10];  // (128,)
  float* out = (float*)d_out;

  char* ws = (char*)d_ws;
  _Float16* vT     = (_Float16*)(ws + OFF_VT);
  float*    sqv    = (float*)(ws + OFF_SQ);
  float*    skv    = (float*)(ws + OFF_SK);
  float*    wq_eff = (float*)(ws + OFF_WQ);
  float*    wk_eff = (float*)(ws + OFF_WK);
  float*    biases = (float*)(ws + OFF_BIAS);
  _Float16* wTv    = (_Float16*)(ws + OFF_WTV);
  _Float16* owT    = (_Float16*)(ws + OFF_OWT);

  prep_kernel<<<1, 256, 0, stream>>>(qw, qb, kvw, kvb, scw, scb, ow,
                                     wq_eff, wk_eff, biases, wTv, owT);
  rowdot_kernel<<<(Bq*Mq + 255)/256, 256, 0, stream>>>(machfeat, wq_eff, biases, 0, sqv, Bq*Mq);
  rowdot_kernel<<<(Bq*Cq + 255)/256, 256, 0, stream>>>(opfeat,   wk_eff, biases, 1, skv, Bq*Cq);
  vproj_kernel<<<(Bq*Cq)/16, 128, 0, stream>>>(opfeat, wTv, kvb, vT);
  attn_kernel<<<Bq*(Mq/16), 128, 0, stream>>>(mask, sqv, skv, vT, owT, ob, out);
}
